// SQEmbedding_67242007986790
// MI455X (gfx1250) — matrix operand order repaired and checked by reading, not compiled, not run on hardware
//
#include <hip/hip_runtime.h>
#include <hip/hip_bf16.h>
#include <math.h>

// ---------------------------------------------------------------------------
// SQEmbedding fused forward for MI455X (gfx1250, wave32, WMMA)
//   logits GEMM  : x(16x128) @ E^T(128x1024)  via bf16 hi/lo split WMMA (3x)
//   softmax etc. : fused in LDS (per-block 16x1024 tile), u streamed once
//   quant GEMM   : enc(16x1024) @ E(1024x128) via bf16 hi/lo split WMMA (3x)
// ---------------------------------------------------------------------------

typedef __bf16 bf16_t;
typedef __attribute__((ext_vector_type(16))) __bf16 v16bf;
typedef __attribute__((ext_vector_type(8)))  __bf16 v8bf;
typedef __attribute__((ext_vector_type(8)))  float  v8f;

#define NTOK 32768
#define MCB  1024
#define DDIM 128
#define RPB  16      // rows per block
#define LPITCH 1032  // padded f32 pitch for 1024-wide LDS tiles

__device__ __forceinline__ float wave_sum(float v) {
#pragma unroll
  for (int o = 16; o > 0; o >>= 1) v += __shfl_xor(v, o, 32);
  return v;
}
__device__ __forceinline__ float wave_max(float v) {
#pragma unroll
  for (int o = 16; o > 0; o >>= 1) v = fmaxf(v, __shfl_xor(v, o, 32));
  return v;
}
__device__ __forceinline__ v16bf cat8(v8bf a, v8bf b) {
  v16bf r;
#pragma unroll
  for (int i = 0; i < 8; ++i) { r[i] = a[i]; r[8 + i] = b[i]; }
  return r;
}

// ---- prep: E -> bf16 hi/lo (row-major + transposed) -----------------------
__global__ void k_prep_embed(const float* __restrict__ E,
                             bf16_t* __restrict__ e_hi, bf16_t* __restrict__ e_lo,
                             bf16_t* __restrict__ e_hiT, bf16_t* __restrict__ e_loT) {
  int t = blockIdx.x * blockDim.x + threadIdx.x;  // 0 .. M*D-1
  int m = t >> 7, d = t & 127;
  float v = E[t];
  bf16_t h  = (bf16_t)v;
  bf16_t lo = (bf16_t)(v - (float)h);
  e_hi[t] = h;  e_lo[t] = lo;
  e_hiT[d * MCB + m] = h;  e_loT[d * MCB + m] = lo;
}

// ---- prep: ||e||^2, zero histogram + loss slot ----------------------------
__global__ void k_prep_misc(const float* __restrict__ E, float* __restrict__ e2,
                            float* __restrict__ counts, float* __restrict__ loss_slot) {
  int m = blockIdx.x * blockDim.x + threadIdx.x;
  if (m < MCB) {
    const float4* p = (const float4*)(E + (size_t)m * DDIM);
    float s = 0.f;
#pragma unroll
    for (int i = 0; i < DDIM / 4; ++i) {
      float4 v = p[i];
      s += v.x * v.x + v.y * v.y + v.z * v.z + v.w * v.w;
    }
    e2[m] = s;
    counts[m] = 0.f;
    if (m == 0) *loss_slot = 0.f;
  }
}

// ---- main fused kernel ----------------------------------------------------
__global__ __launch_bounds__(256) void sq_main(
    const float* __restrict__ x, const float* __restrict__ lvq,
    const float* __restrict__ u,
    const bf16_t* __restrict__ e_hi, const bf16_t* __restrict__ e_lo,
    const bf16_t* __restrict__ e_hiT, const bf16_t* __restrict__ e_loT,
    const float* __restrict__ e2, float* __restrict__ counts,
    float* __restrict__ out) {
  extern __shared__ __align__(16) unsigned char smem_raw[];
  float*  s_logits = (float*)smem_raw;                 // 16 x 1032 f32
  float*  s_ez     = s_logits + RPB * LPITCH;          // 16 x 1032 f32
  float*  s_x      = s_ez + RPB * LPITCH;              // 16 x 128  f32
  bf16_t* s_xhi    = (bf16_t*)(s_x + RPB * DDIM);      // 16 x 128  bf16
  bf16_t* s_xlo    = s_xhi + RPB * DDIM;               // 16 x 128  bf16
  float*  s_x2     = (float*)(s_xlo + RPB * DDIM);     // 16
  float*  s_rcpS   = s_x2 + RPB;                       // 16
  float*  s_acc    = s_rcpS + RPB;                     // [0]=entropy [1]=sq

  const int tid   = threadIdx.x;
  const int lane  = tid & 31;
  const int wv    = tid >> 5;      // 8 waves
  const int row0  = blockIdx.x * RPB;
  const int arow  = lane & 15;
  const int ahalf = lane >> 4;
  const float prec = expf(-lvq[0]);

  if (tid == 0) { s_acc[0] = 0.f; s_acc[1] = 0.f; }

  // -- stage x block, split into bf16 hi/lo --------------------------------
  {
    const float4* gx = (const float4*)(x + (size_t)row0 * DDIM);
    float4* sx4 = (float4*)s_x;
#pragma unroll
    for (int i = 0; i < 2; ++i) {
      int idx = tid + i * 256;           // 512 float4 total
      float4 v = gx[idx];
      sx4[idx] = v;
      float vals[4] = {v.x, v.y, v.z, v.w};
      int e = idx * 4;
#pragma unroll
      for (int j = 0; j < 4; ++j) {
        bf16_t h = (bf16_t)vals[j];
        s_xhi[e + j] = h;
        s_xlo[e + j] = (bf16_t)(vals[j] - (float)h);
      }
    }
  }
  __syncthreads();

  // -- ||x||^2 per row (2 rows per wave) -----------------------------------
#pragma unroll
  for (int rr = wv * 2; rr <= wv * 2 + 1; ++rr) {
    float s = 0.f;
#pragma unroll
    for (int d = lane; d < DDIM; d += 32) { float v = s_x[rr * DDIM + d]; s += v * v; }
    s = wave_sum(s);
    if (lane == 0) s_x2[rr] = s;
  }
  __syncthreads();

  // -- phase 1: logits = prec*x@E^T - 0.5*prec*(x2+e2) into LDS ------------
  v16bf Ahi[4], Alo[4];
#pragma unroll
  for (int kb = 0; kb < 4; ++kb) {       // K = kb*32 .. +31
    int k0 = kb * 32 + ahalf * 16;
    v8bf h1 = *(const v8bf*)&s_xhi[arow * DDIM + k0];
    v8bf h2 = *(const v8bf*)&s_xhi[arow * DDIM + k0 + 8];
    v8bf l1 = *(const v8bf*)&s_xlo[arow * DDIM + k0];
    v8bf l2 = *(const v8bf*)&s_xlo[arow * DDIM + k0 + 8];
    Ahi[kb] = cat8(h1, h2);
    Alo[kb] = cat8(l1, l2);
  }
  float xr[8];
#pragma unroll
  for (int j = 0; j < 8; ++j) xr[j] = s_x2[ahalf * 8 + j];

#pragma unroll 1
  for (int i = 0; i < 8; ++i) {          // 8 column tiles per wave
    int col = (wv * 8 + i) * 16 + arow;
    float e2c = e2[col];
    v8f c = {0.f, 0.f, 0.f, 0.f, 0.f, 0.f, 0.f, 0.f};
#pragma unroll
    for (int kb = 0; kb < 4; ++kb) {
      const v8bf* bh = (const v8bf*)&e_hi[(size_t)col * DDIM + kb * 32 + ahalf * 16];
      const v8bf* bl = (const v8bf*)&e_lo[(size_t)col * DDIM + kb * 32 + ahalf * 16];
      v16bf Bhi = cat8(bh[0], bh[1]);
      v16bf Blo = cat8(bl[0], bl[1]);
      c = __builtin_amdgcn_wmma_f32_16x16x32_bf16(false, Ahi[kb], false, Bhi, (short)0, c, false, false);
      c = __builtin_amdgcn_wmma_f32_16x16x32_bf16(false, Ahi[kb], false, Blo, (short)0, c, false, false);
      c = __builtin_amdgcn_wmma_f32_16x16x32_bf16(false, Alo[kb], false, Bhi, (short)0, c, false, false);
    }
#pragma unroll
    for (int j = 0; j < 8; ++j) {
      int row = ahalf * 8 + j;
      s_logits[row * LPITCH + col] = prec * c[j] - 0.5f * prec * (xr[j] + e2c);
    }
  }
  __syncthreads();

  // -- phase 2: per-row argmax, gumbel-softmax, entropy --------------------
  const float EPS = 1.1920929e-7f;
#pragma unroll 1
  for (int rr = wv * 2; rr <= wv * 2 + 1; ++rr) {
    size_t ubase = (size_t)(row0 + rr) * MCB;
    float bm = -3.4e38f; int bidx = 0; float mz = -3.4e38f;
#pragma unroll 1
    for (int cidx = lane; cidx < MCB; cidx += 32) {
      float l = s_logits[rr * LPITCH + cidx];
      float uu = u[ubase + cidx];
      uu = fminf(fmaxf(uu, EPS), 1.0f - EPS);
      float g = -logf(-logf(uu));
      float z = l + g;                      // TAU = 1
      s_ez[rr * LPITCH + cidx] = z;
      if (l > bm) { bm = l; bidx = cidx; }
      mz = fmaxf(mz, z);
    }
#pragma unroll
    for (int o = 16; o > 0; o >>= 1) {      // argmax: first-index tiebreak
      float ov = __shfl_xor(bm, o, 32);
      int   oi = __shfl_xor(bidx, o, 32);
      if (ov > bm || (ov == bm && oi < bidx)) { bm = ov; bidx = oi; }
    }
    mz = wave_max(mz);
    float Sl = 0.f, T = 0.f, Sz = 0.f;
#pragma unroll 1
    for (int cidx = lane; cidx < MCB; cidx += 32) {
      float dl = s_logits[rr * LPITCH + cidx] - bm;
      float el = expf(dl);
      Sl += el; T += el * dl;
      float ez = expf(s_ez[rr * LPITCH + cidx] - mz);
      Sz += ez;
      s_ez[rr * LPITCH + cidx] = ez;        // unnormalized encodings
    }
    Sl = wave_sum(Sl); T = wave_sum(T); Sz = wave_sum(Sz);
    if (lane == 0) {
      s_rcpS[rr] = 1.0f / Sz;
      atomicAdd(&s_acc[0], T / Sl - logf(Sl));       // sum probs*log_probs
      out[(size_t)NTOK * DDIM + (row0 + rr)] = (float)bidx;
      atomicAdd(&counts[bidx], 1.0f);
    }
  }
  __syncthreads();

  // -- phase 3: quantized = enc @ E (K=1024), + MSE ------------------------
  {
    int d = wv * 16 + arow;
    v8f c = {0.f, 0.f, 0.f, 0.f, 0.f, 0.f, 0.f, 0.f};
#pragma unroll 1
    for (int kb = 0; kb < 32; ++kb) {
      int k0 = kb * 32 + ahalf * 16;
      const float4* pa = (const float4*)&s_ez[arow * LPITCH + k0];
      float4 f0 = pa[0], f1 = pa[1], f2 = pa[2], f3 = pa[3];
      float av[16] = {f0.x, f0.y, f0.z, f0.w, f1.x, f1.y, f1.z, f1.w,
                      f2.x, f2.y, f2.z, f2.w, f3.x, f3.y, f3.z, f3.w};
      v16bf ahi, alo;
#pragma unroll
      for (int i = 0; i < 16; ++i) {
        bf16_t h = (bf16_t)av[i];
        ahi[i] = h;
        alo[i] = (bf16_t)(av[i] - (float)h);
      }
      const v8bf* bh = (const v8bf*)&e_hiT[(size_t)d * MCB + kb * 32 + ahalf * 16];
      const v8bf* bl = (const v8bf*)&e_loT[(size_t)d * MCB + kb * 32 + ahalf * 16];
      v16bf Bhi = cat8(bh[0], bh[1]);
      v16bf Blo = cat8(bl[0], bl[1]);
      c = __builtin_amdgcn_wmma_f32_16x16x32_bf16(false, ahi, false, Bhi, (short)0, c, false, false);
      c = __builtin_amdgcn_wmma_f32_16x16x32_bf16(false, ahi, false, Blo, (short)0, c, false, false);
      c = __builtin_amdgcn_wmma_f32_16x16x32_bf16(false, alo, false, Bhi, (short)0, c, false, false);
    }
    float acc = 0.f;
#pragma unroll
    for (int j = 0; j < 8; ++j) {
      int row = ahalf * 8 + j;
      float q = c[j] * s_rcpS[row];
      out[(size_t)(row0 + row) * DDIM + d] = q;
      float dx = s_x[row * DDIM + d] - q;
      acc += dx * dx;
    }
    acc = wave_sum(acc);
    if (lane == 0) atomicAdd(&s_acc[1], acc);
  }
  __syncthreads();
  if (tid == 0)
    atomicAdd(out + (size_t)NTOK * DDIM + NTOK, 0.5f * prec * s_acc[1] + s_acc[0]);
}

// ---- perplexity from histogram --------------------------------------------
__global__ void k_perp(const float* __restrict__ counts, float* __restrict__ out_perp) {
  __shared__ float red[32];
  int t = threadIdx.x;  // 1024 threads
  float p = counts[t] * (1.0f / (float)NTOK);
  float v = -p * logf(p + 1e-10f);
  v = wave_sum(v);
  if ((t & 31) == 0) red[t >> 5] = v;
  __syncthreads();
  if (t < 32) {
    float s = red[t];
    s = wave_sum(s);
    if (t == 0) *out_perp = expf(s);
  }
}

// ---------------------------------------------------------------------------
extern "C" void kernel_launch(void* const* d_in, const int* in_sizes, int n_in,
                              void* d_out, int out_size, void* d_ws, size_t ws_size,
                              hipStream_t stream) {
  const float* x   = (const float*)d_in[0];
  const float* E   = (const float*)d_in[1];
  const float* lvq = (const float*)d_in[2];
  const float* u   = (const float*)d_in[3];
  float* out = (float*)d_out;

  // workspace layout (~1.03 MB)
  float*  counts = (float*)d_ws;                 // 1024 f32
  float*  e2     = counts + MCB;                 // 1024 f32
  bf16_t* e_hi   = (bf16_t*)(e2 + MCB);          // M*D bf16
  bf16_t* e_lo   = e_hi + MCB * DDIM;
  bf16_t* e_hiT  = e_lo + MCB * DDIM;            // D*M bf16 (transposed)
  bf16_t* e_loT  = e_hiT + MCB * DDIM;

  k_prep_embed<<<(MCB * DDIM) / 256, 256, 0, stream>>>(E, e_hi, e_lo, e_hiT, e_loT);
  k_prep_misc<<<MCB / 256, 256, 0, stream>>>(E, e2, counts,
                                             out + (size_t)NTOK * DDIM + NTOK);

  size_t smem = (size_t)(RPB * LPITCH * 2 + RPB * DDIM) * sizeof(float)
              + (size_t)RPB * DDIM * 2 * sizeof(bf16_t)
              + (RPB + RPB + 2) * sizeof(float) + 16;
  sq_main<<<NTOK / RPB, 256, smem, stream>>>(x, lvq, u, e_hi, e_lo, e_hiT, e_loT,
                                             e2, counts, out);
  k_perp<<<1, MCB, 0, stream>>>(counts, out + (size_t)NTOK * DDIM + NTOK + 1);
}
